// SelfAttention_19078244729546
// MI455X (gfx1250) — matrix operand order repaired and checked by reading, not compiled, not run on hardware
//
#include <hip/hip_runtime.h>
#include <cmath>

// ---------------------------------------------------------------------------
// CDNA5 (gfx1250) WMMA types
// ---------------------------------------------------------------------------
typedef __attribute__((ext_vector_type(16))) _Float16 v16h;
typedef __attribute__((ext_vector_type(8)))  _Float16 v8h;
typedef __attribute__((ext_vector_type(4)))  _Float16 v4h;
typedef __attribute__((ext_vector_type(8)))  float    v8f;
typedef __attribute__((ext_vector_type(4)))  int      v4i;

#define BM 128
#define BN 128
#define BK 32
#define LDSW 40   // 32 halves + 16B pad per row

#define GAS __attribute__((address_space(1)))
#define LAS __attribute__((address_space(3)))

#if __has_builtin(__builtin_amdgcn_global_load_async_to_lds_b128)
#define USE_ASYNC 1
#else
#define USE_ASYNC 0
#endif

// ---------------------------------------------------------------------------
// All-NT f16 GEMM: C[M,N] = act( A[M,K] @ B[N,K]^T + bias )
//   A, B: f16 row-major.  bias: f32 or null.
//   OUT == 0 : C f32 row-major
//   OUT == 1 : C f16 row-major
//   OUT == 2 : C f16 transposed (C[col*ldc + row], for v^T)
//   ACT: 0 none / 1 relu / 2 tanh
// 256 threads = 8 wave32, 4x2 wave grid over 128x128 tile; wave owns 32x64
// -> 8 WMMA per K-step. Double-buffered LDS, one barrier per K-step.
// Staging: pure 2x16B copies per operand per thread; uses
// GLOBAL_LOAD_ASYNC_TO_LDS_B128 (ASYNCcnt) when available.
// ---------------------------------------------------------------------------
template <int OUT, int ACT>
__launch_bounds__(256, 1)
__global__ void gemm_f16wmma(const _Float16* __restrict__ A,
                             const _Float16* __restrict__ B,
                             const float* __restrict__ bias,
                             void* __restrict__ Cv,
                             int M, int N, int K,
                             int lda, int ldb, int ldc)
{
    __shared__ __align__(16) _Float16 As[2][BM * LDSW];
    __shared__ __align__(16) _Float16 Bs[2][BN * LDSW];

    const int t    = threadIdx.x;
    const int lane = t & 31;
    const int wave = t >> 5;        // 0..7
    const int wm   = wave & 3;      // 32-row slice
    const int wn   = wave >> 2;     // 64-col slice (0..1)

    const int m0 = blockIdx.y * BM;
    const int n0 = blockIdx.x * BN;

    v8f acc[2][4] = {};

    // staging: 16 consecutive halves (32B) per thread per operand
    const int srow = t >> 1;             // 0..127
    const int scol = (t & 1) * 16;       // 0 or 16

    const _Float16* aptr = A + (size_t)(m0 + srow) * lda + scol;
    const _Float16* bptr = B + (size_t)(n0 + srow) * ldb + scol;
    _Float16* asl[2] = { &As[0][srow * LDSW + scol], &As[1][srow * LDSW + scol] };
    _Float16* bsl[2] = { &Bs[0][srow * LDSW + scol], &Bs[1][srow * LDSW + scol] };

    auto issue_copy = [&](int buf) {
#if USE_ASYNC
        __builtin_amdgcn_global_load_async_to_lds_b128(
            (GAS v4i*)aptr, (LAS v4i*)asl[buf], 0, 0);
        __builtin_amdgcn_global_load_async_to_lds_b128(
            (GAS v4i*)aptr, (LAS v4i*)asl[buf], 16, 0);
        __builtin_amdgcn_global_load_async_to_lds_b128(
            (GAS v4i*)bptr, (LAS v4i*)bsl[buf], 0, 0);
        __builtin_amdgcn_global_load_async_to_lds_b128(
            (GAS v4i*)bptr, (LAS v4i*)bsl[buf], 16, 0);
#else
        uint4 ra0 = *(const uint4*)(aptr);
        uint4 ra1 = *(const uint4*)(aptr + 8);
        uint4 rb0 = *(const uint4*)(bptr);
        uint4 rb1 = *(const uint4*)(bptr + 8);
        *(uint4*)(asl[buf])     = ra0;
        *(uint4*)(asl[buf] + 8) = ra1;
        *(uint4*)(bsl[buf])     = rb0;
        *(uint4*)(bsl[buf] + 8) = rb1;
#endif
    };
    auto wait_copy = []() {
#if USE_ASYNC
#if __has_builtin(__builtin_amdgcn_s_wait_asynccnt)
        __builtin_amdgcn_s_wait_asynccnt(0);
#else
        asm volatile("s_wait_asynccnt 0x0" ::: "memory");
#endif
#endif
    };

    issue_copy(0);
    wait_copy();
    __syncthreads();

    const int half = lane >> 4;
    const int l16  = lane & 15;
    const int ksteps = K / BK;

    for (int ks = 0; ks < ksteps; ++ks) {
        const int buf   = ks & 1;
        const bool more = (ks + 1 < ksteps);

        // ---- issue next K-tile's copies (overlap with WMMAs) ----
        if (more) {
            aptr += BK;
            bptr += BK;
            issue_copy(buf ^ 1);
            __builtin_prefetch(aptr + BK, 0, 3);   // global_prefetch_b8
        }

        // ---- fragments per ISA wave32 16-bit layouts + 8 WMMAs ----
        union { v16h v; v8h h[2]; } af[2];
        #pragma unroll
        for (int mi = 0; mi < 2; ++mi) {
            // A 16x32: lanes 0-15 K[0..7]+K[16..23]; lanes 16-31 K[8..15]+K[24..31]
            const _Float16* ap = &As[buf][(wm * 32 + mi * 16 + l16) * LDSW + half * 8];
            af[mi].h[0] = *(const v8h*)(ap);
            af[mi].h[1] = *(const v8h*)(ap + 16);
        }
        #pragma unroll
        for (int ni = 0; ni < 4; ++ni) {
            // B 32x16: lane n = col n K[0..7]+K[16..23]; lane n+16 = col n K[8..15]+K[24..31]
            const _Float16* bp = &Bs[buf][(wn * 64 + ni * 16 + l16) * LDSW + half * 8];
            union { v16h v; v8h h[2]; } bf;
            bf.h[0] = *(const v8h*)(bp);
            bf.h[1] = *(const v8h*)(bp + 16);
            acc[0][ni] = __builtin_amdgcn_wmma_f32_16x16x32_f16(
                false, af[0].v, false, bf.v, (short)0, acc[0][ni], false, false);
            acc[1][ni] = __builtin_amdgcn_wmma_f32_16x16x32_f16(
                false, af[1].v, false, bf.v, (short)0, acc[1][ni], false, false);
        }

        if (more) {
            wait_copy();
            __syncthreads();
        }
    }

    // ---- epilogue: bias + activation ----
    #pragma unroll
    for (int mi = 0; mi < 2; ++mi) {
        const int row_base = m0 + wm * 32 + mi * 16 + half * 8;
        #pragma unroll
        for (int ni = 0; ni < 4; ++ni) {
            const int col = n0 + wn * 64 + ni * 16 + l16;
            const float bv = bias ? bias[col] : 0.0f;
            #pragma unroll
            for (int r = 0; r < 8; ++r) {
                float v = acc[mi][ni][r] + bv;
                if (ACT == 1)      v = fmaxf(v, 0.0f);
                else if (ACT == 2) v = tanhf(v);
                const int row = row_base + r;
                if (OUT == 0)
                    ((float*)Cv)[(size_t)row * ldc + col] = v;
                else if (OUT == 1)
                    ((_Float16*)Cv)[(size_t)row * ldc + col] = (_Float16)v;
                else
                    ((_Float16*)Cv)[(size_t)col * ldc + row] = (_Float16)v;
            }
        }
    }
}

// ---------------------------------------------------------------------------
// x: f32 -> f16 (vectorized, n must be multiple of 4)
// ---------------------------------------------------------------------------
__global__ void cvt_f32_f16(const float* __restrict__ in,
                            _Float16* __restrict__ out, int n4)
{
    int i = blockIdx.x * 256 + threadIdx.x;
    if (i < n4) {
        float4 v = ((const float4*)in)[i];
        v4h o = { (_Float16)v.x, (_Float16)v.y, (_Float16)v.z, (_Float16)v.w };
        ((v4h*)out)[i] = o;
    }
}

// ---------------------------------------------------------------------------
// Wt[n*K + k] = (f16) W[k*N + n]   (32x32 LDS tile transpose, 256 threads)
// ---------------------------------------------------------------------------
__global__ void transpose_w_f16(const float* __restrict__ W,
                                _Float16* __restrict__ Wt, int K, int N)
{
    __shared__ float tile[32][33];
    const int tx = threadIdx.x & 31;
    const int ty = threadIdx.x >> 5;          // 0..7
    const int bx = blockIdx.x * 32;           // N base
    const int by = blockIdx.y * 32;           // K base
    #pragma unroll
    for (int j = 0; j < 32; j += 8)
        tile[ty + j][tx] = W[(size_t)(by + ty + j) * N + bx + tx];
    __syncthreads();
    #pragma unroll
    for (int j = 0; j < 32; j += 8)
        Wt[(size_t)(bx + ty + j) * K + by + tx] = (_Float16)tile[tx][ty + j];
}

// ---------------------------------------------------------------------------
// Row softmax: read f32 scores, write f16 probabilities.
// One 256-thread block (8 wave32) per row.
// ---------------------------------------------------------------------------
__global__ void softmax_rows_f16(const float* __restrict__ S,
                                 _Float16* __restrict__ P, int ncols)
{
    const int t = threadIdx.x;
    const float* sp = S + (size_t)blockIdx.x * ncols;
    _Float16*    pp = P + (size_t)blockIdx.x * ncols;
    __shared__ float red[8];

    float m = -3.4e38f;
    for (int i = t; i < ncols; i += 256) m = fmaxf(m, sp[i]);
    #pragma unroll
    for (int off = 16; off; off >>= 1) m = fmaxf(m, __shfl_xor(m, off, 32));
    if ((t & 31) == 0) red[t >> 5] = m;
    __syncthreads();
    float mr = red[0];
    #pragma unroll
    for (int j = 1; j < 8; ++j) mr = fmaxf(mr, red[j]);
    __syncthreads();

    float s = 0.0f;
    for (int i = t; i < ncols; i += 256) {
        float e = __expf(sp[i] - mr);
        pp[i] = (_Float16)e;
        s += e;
    }
    #pragma unroll
    for (int off = 16; off; off >>= 1) s += __shfl_xor(s, off, 32);
    if ((t & 31) == 0) red[t >> 5] = s;
    __syncthreads();
    float sr = 0.0f;
    #pragma unroll
    for (int j = 0; j < 8; ++j) sr += red[j];
    const float inv = 1.0f / sr;
    for (int i = t; i < ncols; i += 256) pp[i] = (_Float16)((float)pp[i] * inv);
}

// ---------------------------------------------------------------------------
// Driver. Workspace (halves unless noted):
//   xh | hsh | qh | kh | vT(HxM) | avh : 6 x 32 MiB
//   W0t..W1t : 5 x 2 MiB (f16 [N,K])
//   sc : T*T f32 (16 MiB) | Ph : T*T f16 (8 MiB)      total ~226 MiB
// ---------------------------------------------------------------------------
extern "C" void kernel_launch(void* const* d_in, const int* in_sizes, int n_in,
                              void* d_out, int out_size, void* d_ws, size_t ws_size,
                              hipStream_t stream)
{
    const float* x  = (const float*)d_in[0];
    const float* W0 = (const float*)d_in[1];
    const float* b0 = (const float*)d_in[2];
    const float* Wq = (const float*)d_in[3];
    const float* bq = (const float*)d_in[4];
    const float* Wk = (const float*)d_in[5];
    const float* bk = (const float*)d_in[6];
    const float* Wv = (const float*)d_in[7];
    const float* bv = (const float*)d_in[8];
    const float* W1 = (const float*)d_in[9];
    const float* b1 = (const float*)d_in[10];
    float* out = (float*)d_out;

    const int Bb = 8, T = 2048, H = 1024;
    const int M = Bb * T;                    // 16384
    const size_t actE = (size_t)M * H;

    char* w = (char*)d_ws;
    _Float16* xh  = (_Float16*)w; w += actE * 2;
    _Float16* hsh = (_Float16*)w; w += actE * 2;
    _Float16* qh  = (_Float16*)w; w += actE * 2;
    _Float16* kh  = (_Float16*)w; w += actE * 2;
    _Float16* vT  = (_Float16*)w; w += actE * 2;     // [H, M]
    _Float16* avh = (_Float16*)w; w += actE * 2;
    _Float16* W0t = (_Float16*)w; w += (size_t)H * H * 2;
    _Float16* Wqt = (_Float16*)w; w += (size_t)H * H * 2;
    _Float16* Wkt = (_Float16*)w; w += (size_t)H * H * 2;
    _Float16* Wvt = (_Float16*)w; w += (size_t)H * H * 2;
    _Float16* W1t = (_Float16*)w; w += (size_t)H * H * 2;
    float*    sc  = (float*)w;    w += (size_t)T * T * 4;
    _Float16* Ph  = (_Float16*)w;

    dim3 blk(256);

    // ---- prep: f16 conversion + weight transposes ----
    cvt_f32_f16<<<(int)(actE / 4 / 256), blk, 0, stream>>>(x, xh, (int)(actE / 4));
    dim3 gT(H / 32, H / 32);
    transpose_w_f16<<<gT, blk, 0, stream>>>(W0, W0t, H, H);
    transpose_w_f16<<<gT, blk, 0, stream>>>(Wq, Wqt, H, H);
    transpose_w_f16<<<gT, blk, 0, stream>>>(Wk, Wkt, H, H);
    transpose_w_f16<<<gT, blk, 0, stream>>>(Wv, Wvt, H, H);
    transpose_w_f16<<<gT, blk, 0, stream>>>(W1, W1t, H, H);

    dim3 gA(H / BN, M / BM);                 // 8 x 128

    // hs = relu(x @ W0 + b0)  -> f16
    gemm_f16wmma<1, 1><<<gA, blk, 0, stream>>>(xh, W0t, b0, hsh, M, H, H, H, H, H);
    // q/k = tanh(hs @ W + b) -> f16 row-major ; v -> f16 transposed [H, M]
    gemm_f16wmma<1, 2><<<gA, blk, 0, stream>>>(hsh, Wqt, bq, qh, M, H, H, H, H, H);
    gemm_f16wmma<1, 2><<<gA, blk, 0, stream>>>(hsh, Wkt, bk, kh, M, H, H, H, H, H);
    gemm_f16wmma<2, 2><<<gA, blk, 0, stream>>>(hsh, Wvt, bv, vT, M, H, H, H, H, M);

    // per-batch attention
    for (int b = 0; b < Bb; ++b) {
        const size_t off = (size_t)b * T * H;
        // sc = q_b @ k_b^T  (f32 out)
        gemm_f16wmma<0, 0><<<dim3(T / BN, T / BM), blk, 0, stream>>>(
            qh + off, kh + off, nullptr, sc, T, T, H, H, H, T);
        softmax_rows_f16<<<T, blk, 0, stream>>>(sc, Ph, T);
        // av_b = P @ v_b : B = vT[:, b*T .. ) as [N=H, K=T] with ldb = M
        gemm_f16wmma<1, 0><<<dim3(H / BN, T / BM), blk, 0, stream>>>(
            Ph, vT + (size_t)b * T, nullptr, avh + off, T, H, T, T, M, H);
    }

    // out = relu(av @ W1 + b1) -> f32
    gemm_f16wmma<0, 1><<<gA, blk, 0, stream>>>(avh, W1t, b1, out, M, H, H, H, H, H);
}
